// CAGatedSelfAttention_12524124635218
// MI455X (gfx1250) — matrix operand order repaired and checked by reading, not compiled, not run on hardware
//
#include <hip/hip_runtime.h>
#include <cstdint>
#include <cstddef>

typedef __attribute__((ext_vector_type(16))) _Float16 v16h;
typedef __attribute__((ext_vector_type(8)))  _Float16 v8h;
typedef __attribute__((ext_vector_type(8)))  float    v8f;

#define B_      2
#define C_      256
#define H_      56
#define W_      56
#define N_      (H_*W_)
#define MID_    32
#define INNER_  64
#define HEADS_  8
#define DH_     8
#define EPS_    1e-5f

#define CAT16(a, b) __builtin_shufflevector(a, b, 0,1,2,3,4,5,6,7,8,9,10,11,12,13,14,15)

static __device__ __forceinline__ v8f wmma16(v16h a, v16h b, v8f c) {
    // D = A(16x32 f16) * B(32x16 f16) + C(16x16 f32)
    return __builtin_amdgcn_wmma_f32_16x16x32_f16(
        /*neg_a=*/false, a, /*neg_b=*/false, b,
        /*c_mod=*/(short)0, c, /*reuse_a=*/false, /*reuse_b=*/false);
}

static __device__ __forceinline__ v8h zero8h() {
    v8h z;
#pragma unroll
    for (int e = 0; e < 8; ++e) z[e] = (_Float16)0.f;
    return z;
}

// ---------------------------------------------------------------------------
// 0) one-shot f32 -> f16 weight conversion
// ---------------------------------------------------------------------------
__global__ __launch_bounds__(256) void cvt_kernel(const float* __restrict__ src,
                                                  _Float16* __restrict__ dst, int n) {
    const int i = blockIdx.x*256 + threadIdx.x;
    if (i < n) dst[i] = (_Float16)src[i];
}

// ---------------------------------------------------------------------------
// 1) h_pool / w_pool : mean over W and over H.  grid (C, B), 64 threads.
// ---------------------------------------------------------------------------
__global__ __launch_bounds__(64) void pool_kernel(const float* __restrict__ x,
                                                  float* __restrict__ hpool,
                                                  float* __restrict__ wpool) {
    const int c = blockIdx.x, b = blockIdx.y, t = threadIdx.x;
    const float* p = x + ((size_t)(b*C_ + c))*N_;
    if (t < H_) {
        float s = 0.f;
        for (int w = 0; w < W_; ++w) s += p[t*W_ + w];
        hpool[((size_t)(b*C_ + c))*H_ + t] = s * (1.f/W_);
        float s2 = 0.f;
        for (int h = 0; h < H_; ++h) s2 += p[h*W_ + t];
        wpool[((size_t)(b*C_ + c))*W_ + t] = s2 * (1.f/H_);
    }
}

// ---------------------------------------------------------------------------
// 2) gate conv + BN + hardswish + sigmoid gates.  grid (B), 256 threads.
// ---------------------------------------------------------------------------
__global__ __launch_bounds__(256) void gates_kernel(
    const float* __restrict__ hpool, const float* __restrict__ wpool,
    const float* __restrict__ gcw,  const float* __restrict__ bng,
    const float* __restrict__ bnb,  const float* __restrict__ bnm,
    const float* __restrict__ bnv,  const float* __restrict__ ghw,
    const float* __restrict__ gww,  float* __restrict__ gh,
    float* __restrict__ gw) {
    __shared__ float cc[MID_*112];
    const int b = blockIdx.x, tid = threadIdx.x;
    for (int t = tid; t < MID_*112; t += 256) {
        const int m = t / 112, l = t % 112;
        float s = 0.f;
        if (l < H_) {
            for (int c = 0; c < C_; ++c)
                s += gcw[m*C_ + c] * hpool[((size_t)(b*C_ + c))*H_ + l];
        } else {
            const int w = l - H_;
            for (int c = 0; c < C_; ++c)
                s += gcw[m*C_ + c] * wpool[((size_t)(b*C_ + c))*W_ + w];
        }
        float v = (s - bnm[m]) * rsqrtf(bnv[m] + EPS_) * bng[m] + bnb[m];
        float r = fminf(fmaxf(v + 3.f, 0.f), 6.f);
        cc[m*112 + l] = v * r * (1.f/6.f);
    }
    __syncthreads();
    for (int t = tid; t < C_*H_; t += 256) {
        const int c = t / H_, pos = t % H_;
        float sh = 0.f, sw = 0.f;
        for (int m = 0; m < MID_; ++m) {
            sh += ghw[c*MID_ + m] * cc[m*112 + pos];
            sw += gww[c*MID_ + m] * cc[m*112 + H_ + pos];
        }
        gh[((size_t)(b*C_ + c))*H_ + pos] = 1.f / (1.f + __expf(-sh));
        gw[((size_t)(b*C_ + c))*W_ + pos] = 1.f / (1.f + __expf(-sw));
    }
}

// ---------------------------------------------------------------------------
// 3) gate_spatial -> log bias.  grid (B), 256 threads.
// ---------------------------------------------------------------------------
__global__ __launch_bounds__(256) void bias_kernel(const float* __restrict__ gh,
                                                   const float* __restrict__ gw,
                                                   float* __restrict__ biasv) {
    const int b = blockIdx.x;
    for (int t = threadIdx.x; t < N_; t += 256) {
        const int h = t / W_, w = t % W_;
        float s = 0.f;
        for (int c = 0; c < C_; ++c)
            s += gh[((size_t)(b*C_ + c))*H_ + h] * gw[((size_t)(b*C_ + c))*W_ + w];
        s *= (1.f/C_);
        biasv[(size_t)b*N_ + t] = fmaxf(logf(s), -5.f);
    }
}

// ---------------------------------------------------------------------------
// 4) seq = proj_in @ x (WMMA, M=i orientation) + LayerNorm -> f16.
//    grid (N/16, B), 32 threads (one wave).
// ---------------------------------------------------------------------------
__global__ __launch_bounds__(32) void seq_ln_kernel(
    const float* __restrict__ x, const _Float16* __restrict__ pinH,
    const float* __restrict__ lng, const float* __restrict__ lnb,
    _Float16* __restrict__ seqH) {
    __shared__ _Float16 Xt[16*32];   // [n_local][c_local]
    const int lane = threadIdx.x, hlf = lane >> 4, lcol = lane & 15;
    const int nbase = blockIdx.x * 16, b = blockIdx.y;

    v8f acc[4];
#pragma unroll
    for (int t = 0; t < 4; ++t)
#pragma unroll
        for (int r = 0; r < 8; ++r) acc[t][r] = 0.f;

    for (int kc = 0; kc < C_/32; ++kc) {
        const int c0 = kc * 32;
        const float* xr = x + ((size_t)(b*C_ + c0 + lane))*N_ + nbase;
        float xv[16];
#pragma unroll
        for (int j = 0; j < 16; ++j) xv[j] = xr[j];
#pragma unroll
        for (int j = 0; j < 16; ++j) Xt[j*32 + lane] = (_Float16)xv[j];
        __syncthreads();
        const _Float16* bp = &Xt[lcol*32 + 8*hlf];
        const v8h b0 = *(const v8h*)bp;
        const v8h b1 = *(const v8h*)(bp + 16);
        const v16h bX = CAT16(b0, b1);
#pragma unroll
        for (int t = 0; t < 4; ++t) {
            const _Float16* wr = pinH + ((size_t)(t*16 + lcol))*C_ + c0 + 8*hlf;
            const v8h a0 = *(const v8h*)wr;
            const v8h a1 = *(const v8h*)(wr + 16);
            const v16h aW = CAT16(a0, a1);
            acc[t] = wmma16(aW, bX, acc[t]);
        }
        __syncthreads();
    }
    // LayerNorm over i=64 for column n = nbase + lcol
    float s = 0.f, sq = 0.f;
#pragma unroll
    for (int t = 0; t < 4; ++t)
#pragma unroll
        for (int r = 0; r < 8; ++r) { const float v = acc[t][r]; s += v; sq += v*v; }
    s  += __shfl_xor(s, 16, 32);
    sq += __shfl_xor(sq, 16, 32);
    const float mean = s * (1.f/INNER_);
    const float var  = sq * (1.f/INNER_) - mean*mean;
    const float rstd = rsqrtf(var + EPS_);
    _Float16* outp = seqH + ((size_t)b*N_ + nbase + lcol)*INNER_;
#pragma unroll
    for (int t = 0; t < 4; ++t)
#pragma unroll
        for (int r = 0; r < 8; ++r) {
            const int i = t*16 + r + 8*hlf;
            outp[i] = (_Float16)((acc[t][r] - mean)*rstd*lng[i] + lnb[i]);
        }
}

// ---------------------------------------------------------------------------
// 5) q/k/v = seq @ w.T  (WMMA, K=64).
//    Q,K -> [b][h][n][dh] f16 ;  V -> transposed [b][h][dpad=16][n] f16.
//    grid (N/16, B), 32 threads.
// ---------------------------------------------------------------------------
__global__ __launch_bounds__(32) void qkv_kernel(
    const _Float16* __restrict__ seqH,
    const _Float16* __restrict__ wqH, const _Float16* __restrict__ wkH,
    const _Float16* __restrict__ wvH,
    _Float16* __restrict__ Qh, _Float16* __restrict__ Kh,
    _Float16* __restrict__ Vt) {
    const int lane = threadIdx.x, hlf = lane >> 4, lcol = lane & 15;
    const int nbase = blockIdx.x * 16, b = blockIdx.y;

    v16h aS[2];
    const _Float16* sp = seqH + ((size_t)b*N_ + nbase + lcol)*INNER_;
#pragma unroll
    for (int ks = 0; ks < 2; ++ks) {
        const _Float16* ap = sp + ks*32 + 8*hlf;
        const v8h a0 = *(const v8h*)ap;
        const v8h a1 = *(const v8h*)(ap + 16);
        aS[ks] = CAT16(a0, a1);
    }
    const _Float16* wmat[3] = { wqH, wkH, wvH };
#pragma unroll
    for (int wsel = 0; wsel < 3; ++wsel) {
        const _Float16* wm = wmat[wsel];
#pragma unroll
        for (int jt = 0; jt < 4; ++jt) {
            v8f acc;
#pragma unroll
            for (int r = 0; r < 8; ++r) acc[r] = 0.f;
#pragma unroll
            for (int ks = 0; ks < 2; ++ks) {
                const _Float16* wr =
                    wm + ((size_t)(jt*16 + lcol))*INNER_ + ks*32 + 8*hlf;
                const v8h b0 = *(const v8h*)wr;
                const v8h b1 = *(const v8h*)(wr + 16);
                const v16h bW = CAT16(b0, b1);
                acc = wmma16(aS[ks], bW, acc);
            }
            const int j = jt*16 + lcol, hh = j >> 3, d = j & 7;
#pragma unroll
            for (int r = 0; r < 8; ++r) {
                const int n = nbase + r + 8*hlf;
                if (wsel == 0)
                    Qh[(((size_t)b*HEADS_ + hh)*N_ + n)*DH_ + d] = (_Float16)acc[r];
                else if (wsel == 1)
                    Kh[(((size_t)b*HEADS_ + hh)*N_ + n)*DH_ + d] = (_Float16)acc[r];
                else
                    Vt[((size_t)(b*HEADS_ + hh)*16 + d)*N_ + n] = (_Float16)acc[r];
            }
        }
    }
}

// ---------------------------------------------------------------------------
// 6) Flash attention: one wave per (b, head, 16-query tile).
//    Bias chunks stream through LDS via double-buffered
//    global_load_async_to_lds_b32 (ASYNCcnt pipeline).
//    grid (N/16, HEADS, B), 32 threads.
// ---------------------------------------------------------------------------
__global__ __launch_bounds__(32) void attn_kernel(
    const _Float16* __restrict__ Qh, const _Float16* __restrict__ Kh,
    const _Float16* __restrict__ Vt, const float* __restrict__ biasv,
    _Float16* __restrict__ attnOh) {
    __shared__ _Float16 Plds[16*32];
    __shared__ float    Blds[2][32];

    const int lane = threadIdx.x, hlf = lane >> 4, lcol = lane & 15;
    const int qtile = blockIdx.x, h = blockIdx.y, b = blockIdx.z;
    const int bh = b*HEADS_ + h;
    const int qbase = qtile*16;
    const float scale = 0.3535533905932738f;   // dh^-0.5, dh=8

    const v8h z8 = zero8h();
    v16h aQ = CAT16(z8, z8);
    if (hlf == 0) {
        const v8h q = *(const v8h*)(Qh + ((size_t)bh*N_ + qbase + lane)*DH_);
        aQ = CAT16(q, z8);
    }
    v8f O, zeroC;
    float mrow[8], lrow[8];
#pragma unroll
    for (int r = 0; r < 8; ++r) { O[r] = 0.f; zeroC[r] = 0.f; mrow[r] = -1e30f; lrow[r] = 0.f; }

    const float* bsrc = biasv + (size_t)b*N_ + lane;
    // prologue: async-stage bias chunk 0
    {
        const uint32_t la = (uint32_t)(uintptr_t)(&Blds[0][lane]);
        const uint64_t ga = (uint64_t)(uintptr_t)(bsrc);
        asm volatile("global_load_async_to_lds_b32 %0, %1, off"
                     :: "v"(la), "v"(ga) : "memory");
    }

    const int nIter = N_/32;
    for (int kb = 0; kb < nIter; ++kb) {
        // --- K tile: direct 16-B global loads (lanes >=16 are zero K-rows)
        const _Float16* kp = Kh + ((size_t)bh*N_ + kb*32)*DH_;
        v8h k0 = z8, k1 = z8;
        if (hlf == 0) {
            k0 = *(const v8h*)(kp + lcol*DH_);
            k1 = *(const v8h*)(kp + (16 + lcol)*DH_);
        }
        const v16h bK0 = CAT16(k0, z8);
        const v16h bK1 = CAT16(k1, z8);
        // --- V tile: direct from transposed layout ------------------------
        const _Float16* vrow = Vt + ((size_t)(bh*16 + lcol))*N_ + kb*32 + 8*hlf;
        v8h v0 = *(const v8h*)vrow;
        v8h v1 = *(const v8h*)(vrow + 16);
        if (lcol >= DH_) { v0 = z8; v1 = z8; }
        const v16h bV = CAT16(v0, v1);
        if (kb + 1 < nIter) {               // gfx1250 global_prefetch
            __builtin_prefetch(kp + 32*DH_, 0, 1);
            __builtin_prefetch(vrow + 32, 0, 1);
        }

        v8f S0 = wmma16(aQ, bK0, zeroC);
        v8f S1 = wmma16(aQ, bK1, zeroC);

        // --- double-buffered async bias pipeline --------------------------
        if (kb + 1 < nIter) {
            const uint32_t la = (uint32_t)(uintptr_t)(&Blds[(kb + 1) & 1][lane]);
            const uint64_t ga = (uint64_t)(uintptr_t)(bsrc + (kb + 1)*32);
            asm volatile("global_load_async_to_lds_b32 %0, %1, off"
                         :: "v"(la), "v"(ga) : "memory");
            asm volatile("s_wait_asynccnt 0x1" ::: "memory");
        } else {
            asm volatile("s_wait_asynccnt 0x0" ::: "memory");
        }
        const float bias0 = Blds[kb & 1][lcol], bias1 = Blds[kb & 1][16 + lcol];

        float s0[8], s1[8], mc[8];
#pragma unroll
        for (int r = 0; r < 8; ++r) {
            s0[r] = S0[r]*scale + bias0;
            s1[r] = S1[r]*scale + bias1;
            mc[r] = fmaxf(s0[r], s1[r]);
        }
#pragma unroll
        for (int off = 1; off < 16; off <<= 1)
#pragma unroll
            for (int r = 0; r < 8; ++r) mc[r] = fmaxf(mc[r], __shfl_xor(mc[r], off, 32));

        float corr[8], p0[8], p1[8], rs[8];
#pragma unroll
        for (int r = 0; r < 8; ++r) {
            const float mn = fmaxf(mrow[r], mc[r]);
            corr[r] = __expf(mrow[r] - mn);
            mrow[r] = mn;
            p0[r] = __expf(s0[r] - mn);
            p1[r] = __expf(s1[r] - mn);
            rs[r] = p0[r] + p1[r];
        }
#pragma unroll
        for (int off = 1; off < 16; off <<= 1)
#pragma unroll
            for (int r = 0; r < 8; ++r) rs[r] += __shfl_xor(rs[r], off, 32);
#pragma unroll
        for (int r = 0; r < 8; ++r) {
            lrow[r] = lrow[r]*corr[r] + rs[r];
            O[r] *= corr[r];
            Plds[(r + 8*hlf)*32 + lcol]      = (_Float16)p0[r];
            Plds[(r + 8*hlf)*32 + 16 + lcol] = (_Float16)p1[r];
        }
        __syncthreads();
        const _Float16* pp = &Plds[lcol*32 + 8*hlf];
        const v8h pa = *(const v8h*)pp;
        const v8h pb = *(const v8h*)(pp + 16);
        const v16h aP = CAT16(pa, pb);
        O = wmma16(aP, bV, O);
        __syncthreads();
    }
#pragma unroll
    for (int r = 0; r < 8; ++r) {
        if (lcol < DH_) {
            const int n = qbase + r + 8*hlf;
            attnOh[((size_t)b*N_ + n)*INNER_ + h*DH_ + lcol] = (_Float16)(O[r]/lrow[r]);
        }
    }
}

// ---------------------------------------------------------------------------
// 7a) zero channel stat accumulators
// ---------------------------------------------------------------------------
__global__ __launch_bounds__(256) void zero_kernel(float* __restrict__ csum,
                                                   float* __restrict__ csumsq) {
    const int i = blockIdx.x*256 + threadIdx.x;
    if (i < B_*C_) { csum[i] = 0.f; csumsq[i] = 0.f; }
}

// ---------------------------------------------------------------------------
// 7b) proj_out (WMMA, K=64) + per-channel sum/sumsq atomics.
//     grid (N/16, C/16, B), 32 threads.
// ---------------------------------------------------------------------------
__global__ __launch_bounds__(32) void projout_kernel(
    const _Float16* __restrict__ attnOh, const _Float16* __restrict__ powmH,
    float* __restrict__ out0, float* __restrict__ csum,
    float* __restrict__ csumsq) {
    const int lane = threadIdx.x, hlf = lane >> 4, lcol = lane & 15;
    const int nbase = blockIdx.x*16, ct = blockIdx.y, b = blockIdx.z;

    v8f acc;
#pragma unroll
    for (int r = 0; r < 8; ++r) acc[r] = 0.f;
#pragma unroll
    for (int ks = 0; ks < 2; ++ks) {
        const _Float16* ar =
            powmH + ((size_t)(ct*16 + lcol))*INNER_ + ks*32 + 8*hlf;
        const v8h a0 = *(const v8h*)ar;
        const v8h a1 = *(const v8h*)(ar + 16);
        const v16h aA = CAT16(a0, a1);
        const _Float16* br =
            attnOh + ((size_t)b*N_ + nbase + lcol)*INNER_ + ks*32 + 8*hlf;
        const v8h r0 = *(const v8h*)br;
        const v8h r1 = *(const v8h*)(br + 16);
        const v16h bB = CAT16(r0, r1);
        acc = wmma16(aA, bB, acc);
    }
    float rsum[8], rsq[8];
#pragma unroll
    for (int r = 0; r < 8; ++r) {
        const int c = ct*16 + r + 8*hlf;
        out0[((size_t)b*C_ + c)*N_ + nbase + lcol] = acc[r];
        rsum[r] = acc[r];
        rsq[r]  = acc[r]*acc[r];
    }
#pragma unroll
    for (int off = 1; off < 16; off <<= 1)
#pragma unroll
        for (int r = 0; r < 8; ++r) {
            rsum[r] += __shfl_xor(rsum[r], off, 32);
            rsq[r]  += __shfl_xor(rsq[r],  off, 32);
        }
    if (lcol == 0) {
#pragma unroll
        for (int r = 0; r < 8; ++r) {
            const int c = ct*16 + r + 8*hlf;
            atomicAdd(&csum[b*C_ + c],   rsum[r]);
            atomicAdd(&csumsq[b*C_ + c], rsq[r]);
        }
    }
}

// ---------------------------------------------------------------------------
// 7c) group stats.  1 block, 64 threads.
// ---------------------------------------------------------------------------
__global__ __launch_bounds__(64) void gstats_kernel(
    const float* __restrict__ csum, const float* __restrict__ csumsq,
    float* __restrict__ gmean, float* __restrict__ grstd) {
    const int t = threadIdx.x;
    if (t < B_*32) {
        const int b = t / 32, g = t % 32;
        float s = 0.f, sq = 0.f;
        for (int j = 0; j < C_/32; ++j) {
            s  += csum[b*C_ + g*(C_/32) + j];
            sq += csumsq[b*C_ + g*(C_/32) + j];
        }
        const float cnt = (float)((C_/32)*N_);
        const float mean = s / cnt;
        const float var  = sq / cnt - mean*mean;
        gmean[t] = mean;
        grstd[t] = rsqrtf(var + EPS_);
    }
}

// ---------------------------------------------------------------------------
// 7d) GroupNorm apply + residual.
// ---------------------------------------------------------------------------
__global__ __launch_bounds__(256) void final_kernel(
    const float* __restrict__ x, const float* __restrict__ out0,
    const float* __restrict__ gmean, const float* __restrict__ grstd,
    const float* __restrict__ gng, const float* __restrict__ gnb,
    float* __restrict__ out) {
    const size_t idx = (size_t)blockIdx.x*256 + threadIdx.x;
    if (idx < (size_t)B_*C_*N_) {
        const int b = (int)(idx / ((size_t)C_*N_));
        const int c = (int)((idx / N_) % C_);
        const int g = b*32 + c/(C_/32);
        const float v = (out0[idx] - gmean[g])*grstd[g]*gng[c] + gnb[c];
        out[idx] = x[idx] + v;
    }
}

// ---------------------------------------------------------------------------
extern "C" void kernel_launch(void* const* d_in, const int* in_sizes, int n_in,
                              void* d_out, int out_size, void* d_ws, size_t ws_size,
                              hipStream_t stream) {
    (void)in_sizes; (void)n_in; (void)out_size; (void)ws_size;
    const float* x    = (const float*)d_in[0];
    const float* gcw  = (const float*)d_in[1];
    const float* bng  = (const float*)d_in[2];
    const float* bnb  = (const float*)d_in[3];
    const float* bnm  = (const float*)d_in[4];
    const float* bnv  = (const float*)d_in[5];
    const float* ghw  = (const float*)d_in[6];
    const float* gww  = (const float*)d_in[7];
    const float* pin  = (const float*)d_in[8];
    const float* lng  = (const float*)d_in[9];
    const float* lnb  = (const float*)d_in[10];
    const float* wq   = (const float*)d_in[11];
    const float* wk   = (const float*)d_in[12];
    const float* wv   = (const float*)d_in[13];
    const float* powm = (const float*)d_in[14];
    const float* gng  = (const float*)d_in[15];
    const float* gnb  = (const float*)d_in[16];
    float* out = (float*)d_out;

    char* p = (char*)d_ws;
    auto alloc = [&](size_t bytes) -> char* {
        char* r = p; p += (bytes + 255) & ~(size_t)255; return r;
    };
    float*    hpool  = (float*)alloc((size_t)B_*C_*H_*4);
    float*    wpool  = (float*)alloc((size_t)B_*C_*W_*4);
    float*    gh     = (float*)alloc((size_t)B_*C_*H_*4);
    float*    gw     = (float*)alloc((size_t)B_*C_*W_*4);
    float*    biasv  = (float*)alloc((size_t)B_*N_*4);
    _Float16* seqH   = (_Float16*)alloc((size_t)B_*N_*INNER_*2);
    _Float16* Qh     = (_Float16*)alloc((size_t)B_*HEADS_*N_*DH_*2);
    _Float16* Kh     = (_Float16*)alloc((size_t)B_*HEADS_*N_*DH_*2);
    _Float16* Vt     = (_Float16*)alloc((size_t)B_*HEADS_*16*N_*2);  // d padded to 16
    _Float16* attnOh = (_Float16*)alloc((size_t)B_*N_*INNER_*2);
    float*    out0   = (float*)alloc((size_t)B_*C_*N_*4);
    float*    csum   = (float*)alloc((size_t)B_*C_*4);
    float*    csumsq = (float*)alloc((size_t)B_*C_*4);
    float*    gmean  = (float*)alloc((size_t)B_*32*4);
    float*    grstd  = (float*)alloc((size_t)B_*32*4);
    _Float16* pinH   = (_Float16*)alloc((size_t)INNER_*C_*2);
    _Float16* wqH    = (_Float16*)alloc((size_t)INNER_*INNER_*2);
    _Float16* wkH    = (_Float16*)alloc((size_t)INNER_*INNER_*2);
    _Float16* wvH    = (_Float16*)alloc((size_t)INNER_*INNER_*2);
    _Float16* powmH  = (_Float16*)alloc((size_t)C_*INNER_*2);

    // one-shot weight conversions to f16
    cvt_kernel<<<dim3((INNER_*C_ + 255)/256),    256, 0, stream>>>(pin,  pinH,  INNER_*C_);
    cvt_kernel<<<dim3((INNER_*INNER_ + 255)/256),256, 0, stream>>>(wq,   wqH,   INNER_*INNER_);
    cvt_kernel<<<dim3((INNER_*INNER_ + 255)/256),256, 0, stream>>>(wk,   wkH,   INNER_*INNER_);
    cvt_kernel<<<dim3((INNER_*INNER_ + 255)/256),256, 0, stream>>>(wv,   wvH,   INNER_*INNER_);
    cvt_kernel<<<dim3((C_*INNER_ + 255)/256),    256, 0, stream>>>(powm, powmH, C_*INNER_);

    pool_kernel   <<<dim3(C_, B_),            64, 0, stream>>>(x, hpool, wpool);
    gates_kernel  <<<dim3(B_),               256, 0, stream>>>(hpool, wpool, gcw, bng, bnb,
                                                               bnm, bnv, ghw, gww, gh, gw);
    bias_kernel   <<<dim3(B_),               256, 0, stream>>>(gh, gw, biasv);
    seq_ln_kernel <<<dim3(N_/16, B_),         32, 0, stream>>>(x, pinH, lng, lnb, seqH);
    qkv_kernel    <<<dim3(N_/16, B_),         32, 0, stream>>>(seqH, wqH, wkH, wvH, Qh, Kh, Vt);
    attn_kernel   <<<dim3(N_/16, HEADS_, B_), 32, 0, stream>>>(Qh, Kh, Vt, biasv, attnOh);
    zero_kernel   <<<dim3((B_*C_ + 255)/256),256, 0, stream>>>(csum, csumsq);
    projout_kernel<<<dim3(N_/16, C_/16, B_),  32, 0, stream>>>(attnOh, powmH, out0, csum, csumsq);
    gstats_kernel <<<dim3(1),                 64, 0, stream>>>(csum, csumsq, gmean, grstd);
    final_kernel  <<<dim3((int)(((size_t)B_*C_*N_ + 255)/256)), 256, 0, stream>>>(
        x, out0, gmean, grstd, gng, gnb, out);
}
